// SimpleDecoderAttention_18519898980568
// MI455X (gfx1250) — matrix operand order repaired and checked by reading, not compiled, not run on hardware
//
#include <hip/hip_runtime.h>
#include <stdint.h>

// ---------------------------------------------------------------------------
// Types
// ---------------------------------------------------------------------------
typedef __attribute__((ext_vector_type(16))) __bf16 v16bf;
typedef __attribute__((ext_vector_type(8)))  float  v8f;
typedef __attribute__((ext_vector_type(4)))  float  vf4;
typedef __attribute__((ext_vector_type(4)))  __bf16 vbf4;

union Frag { v16bf v; vf4 f4[2]; };   // 32B A/B operand for wmma 16x16x32 bf16

__device__ __forceinline__ __bf16 f2bf(float f) {
  return (__bf16)f;                   // native f32->bf16 convert
}

__device__ __forceinline__ float fast_exp2(float x) {
  return __builtin_amdgcn_exp2f(x);   // v_exp_f32
}

#define EMBED 2048
#define NHEAD 16
#define HDIM  128
#define HALF  64
#define BATCH 2
#define SEQ   2048

// ---------------------------------------------------------------------------
// GEMM: C[M,N] = A[M,K] @ W[K,N] + bias[N]   (f32 in/out, bf16 WMMA compute)
// Block tile 128x256, BK=32, 256 threads = 8 waves (2x4), wave tile 64x64:
// 16 wmma per wave per K-step against 16 ds_load_b128 fragment loads.
// ---------------------------------------------------------------------------
__global__ __launch_bounds__(256) void gemm_bias_kernel(
    const float* __restrict__ A, const float* __restrict__ W,
    const float* __restrict__ bias, float* __restrict__ C,
    int M, int N, int K)
{
  __shared__ __align__(16) __bf16 As[128][32];   // m-major, k-contiguous (8KB)
  __shared__ __align__(16) __bf16 Bs[256][32];   // n-major, k-contiguous (16KB)

  const int tid  = threadIdx.x;
  const int lane = tid & 31;
  const int wave = tid >> 5;
  const int wm   = wave >> 2;          // 0..1  (64 rows each)
  const int wn   = wave & 3;           // 0..3  (64 cols each)
  const int lo   = lane & 15;
  const int hi   = lane >> 4;
  const int bm   = blockIdx.y * 128;
  const int bn   = blockIdx.x * 256;

  v8f acc[4][4];
  const v8f vzero = {0.f,0.f,0.f,0.f,0.f,0.f,0.f,0.f};
  for (int i = 0; i < 4; ++i)
    for (int j = 0; j < 4; ++j) acc[i][j] = vzero;

  for (int k0 = 0; k0 < K; k0 += 32) {
    // --- stage A tile (128x32 f32 -> bf16), rows contiguous in k
    #pragma unroll
    for (int i = 0; i < 4; ++i) {
      int idx = tid + i * 256;               // 0..1023
      int row = idx >> 3;                    // 0..127
      int col = (idx & 7) << 2;              // 0,4,..,28
      vf4 a = *reinterpret_cast<const vf4*>(&A[(size_t)(bm + row) * K + k0 + col]);
      vbf4 p = { f2bf(a.x), f2bf(a.y), f2bf(a.z), f2bf(a.w) };
      *reinterpret_cast<vbf4*>(&As[row][col]) = p;
    }
    // --- stage W tile transposed (32x256 f32 -> Bs[n][k] bf16)
    #pragma unroll
    for (int i = 0; i < 8; ++i) {
      int idx = tid + i * 256;               // 0..2047
      int kr  = idx >> 6;                    // 0..31
      int cn  = (idx & 63) << 2;             // 0..252
      vf4 w = *reinterpret_cast<const vf4*>(&W[(size_t)(k0 + kr) * N + bn + cn]);
      Bs[cn + 0][kr] = f2bf(w.x);
      Bs[cn + 1][kr] = f2bf(w.y);
      Bs[cn + 2][kr] = f2bf(w.z);
      Bs[cn + 3][kr] = f2bf(w.w);
    }
    __syncthreads();

    // --- fragments per ISA 16-bit layouts
    Frag af[4], bf[4];
    #pragma unroll
    for (int i = 0; i < 4; ++i) {
      int row = wm * 64 + i * 16 + lo;
      int kb  = hi * 8;                      // A: k-chunks {kb..kb+7},{kb+16..kb+23}
      af[i].f4[0] = *reinterpret_cast<const vf4*>(&As[row][kb]);
      af[i].f4[1] = *reinterpret_cast<const vf4*>(&As[row][kb + 16]);
    }
    #pragma unroll
    for (int j = 0; j < 4; ++j) {
      int col = wn * 64 + j * 16 + lo;
      int kb  = hi * 8;                      // B: k-chunks {kb..kb+7},{kb+16..kb+23}
      bf[j].f4[0] = *reinterpret_cast<const vf4*>(&Bs[col][kb]);
      bf[j].f4[1] = *reinterpret_cast<const vf4*>(&Bs[col][kb + 16]);
    }
    #pragma unroll
    for (int i = 0; i < 4; ++i)
      #pragma unroll
      for (int j = 0; j < 4; ++j)
        acc[i][j] = __builtin_amdgcn_wmma_f32_16x16x32_bf16(
            false, af[i].v, false, bf[j].v, (short)0, acc[i][j], false, false);
    __syncthreads();
  }

  // --- epilogue: C layout -> global (VGPR r: row r / 8+r per half-wave)
  #pragma unroll
  for (int i = 0; i < 4; ++i)
    #pragma unroll
    for (int j = 0; j < 4; ++j) {
      int col = bn + wn * 64 + j * 16 + lo;
      float bv = bias[col];
      int row0 = bm + wm * 64 + i * 16 + hi * 8;
      float* cp = C + (size_t)row0 * N + col;
      #pragma unroll
      for (int r = 0; r < 8; ++r) {
        *cp = acc[i][j][r] + bv;
        cp += N;
      }
    }
}

// ---------------------------------------------------------------------------
// RoPE (pos clipped to {0,1}; pos==0 is identity) applied to q and k in place
// ---------------------------------------------------------------------------
__global__ __launch_bounds__(256) void rope_kernel(
    float* __restrict__ q, float* __restrict__ k,
    const int* __restrict__ pos_ids)
{
  int idx = blockIdx.x * blockDim.x + threadIdx.x;   // B*S*H*HALF threads
  int j = idx & (HALF - 1);
  int h = (idx >> 6) & (NHEAD - 1);
  int s = (idx >> 10) & (SEQ - 1);
  int b = idx >> 21;
  int p = pos_ids[b * SEQ + s];
  p = p < 0 ? 0 : (p > 1 ? 1 : p);
  if (p == 0) return;
  float ang = __powf(10000.0f, -(float)j * (1.0f / (float)HALF));
  float c = __cosf(ang), sn = __sinf(ang);
  size_t base = (((size_t)b * SEQ + s) * NHEAD + h) * HDIM;
  float qr = q[base + j], qi = q[base + j + HALF];
  q[base + j]        = qr * c - qi * sn;
  q[base + j + HALF] = qi * c + qr * sn;
  float kr = k[base + j], ki = k[base + j + HALF];
  k[base + j]        = kr * c - ki * sn;
  k[base + j + HALF] = ki * c + kr * sn;
}

// ---------------------------------------------------------------------------
// Flash attention: per block = 64 q-rows of one (b,h); 4 waves x 16 rows.
// Key blocks of 32; online softmax in exp2 domain; all matmuls via bf16 WMMA.
// ---------------------------------------------------------------------------
__global__ __launch_bounds__(128) void flash_attn_kernel(
    const float* __restrict__ q, const float* __restrict__ k,
    const float* __restrict__ v, float* __restrict__ o)
{
  __shared__ __align__(16) __bf16 Ks[32][HDIM];    // key-major (B-frag for Q@K^T)
  __shared__ __align__(16) __bf16 Vt[HDIM][32];    // d-major   (B-frag for P@V)
  __shared__ __align__(16) __bf16 Ps[4][16][32];   // per-wave P staging

  const int tid  = threadIdx.x;
  const int lane = tid & 31;
  const int wave = tid >> 5;
  const int lo   = lane & 15;
  const int hi   = lane >> 4;
  const int bh   = blockIdx.y;
  const int b    = bh >> 4;
  const int h    = bh & (NHEAD - 1);
  const int q0   = blockIdx.x * 64;

  const size_t srow    = (size_t)NHEAD * HDIM;
  const size_t bh_base = ((size_t)b * SEQ) * srow + (size_t)h * HDIM;

  // Q fragments held in registers for the whole loop (4 chunks of k=32)
  Frag qf[4];
  {
    int row = q0 + wave * 16 + lo;
    const float* qp = q + bh_base + (size_t)row * srow;
    #pragma unroll
    for (int c = 0; c < 4; ++c) {
      int d0 = c * 32 + hi * 8;
      vf4 x0 = *reinterpret_cast<const vf4*>(qp + d0);
      vf4 x1 = *reinterpret_cast<const vf4*>(qp + d0 + 4);
      vf4 x2 = *reinterpret_cast<const vf4*>(qp + d0 + 16);
      vf4 x3 = *reinterpret_cast<const vf4*>(qp + d0 + 20);
      qf[c].v[0]=f2bf(x0.x);  qf[c].v[1]=f2bf(x0.y);  qf[c].v[2]=f2bf(x0.z);  qf[c].v[3]=f2bf(x0.w);
      qf[c].v[4]=f2bf(x1.x);  qf[c].v[5]=f2bf(x1.y);  qf[c].v[6]=f2bf(x1.z);  qf[c].v[7]=f2bf(x1.w);
      qf[c].v[8]=f2bf(x2.x);  qf[c].v[9]=f2bf(x2.y);  qf[c].v[10]=f2bf(x2.z); qf[c].v[11]=f2bf(x2.w);
      qf[c].v[12]=f2bf(x3.x); qf[c].v[13]=f2bf(x3.y); qf[c].v[14]=f2bf(x3.z); qf[c].v[15]=f2bf(x3.w);
    }
  }

  const v8f vzero = {0.f,0.f,0.f,0.f,0.f,0.f,0.f,0.f};
  v8f   oacc[8];
  float mrow[8], lrow[8];
  #pragma unroll
  for (int t = 0; t < 8; ++t) oacc[t] = vzero;
  #pragma unroll
  for (int r = 0; r < 8; ++r) { mrow[r] = -1e30f; lrow[r] = 0.0f; }

  const float scale2 = 0.08838834764831845f * 1.4426950408889634f; // 1/sqrt(128)*log2e

  for (int kb = 0; kb < SEQ; kb += 32) {
    // ---- cooperative K/V tile staging (f32 -> bf16; V transposed)
    #pragma unroll
    for (int i = 0; i < 8; ++i) {
      int idx = tid + i * 128;               // 0..1023
      int row = idx >> 5;                    // key 0..31
      int c4  = (idx & 31) << 2;             // d 0..124
      size_t g = bh_base + (size_t)(kb + row) * srow + c4;
      vf4 kv = *reinterpret_cast<const vf4*>(k + g);
      vbf4 kp = { f2bf(kv.x), f2bf(kv.y), f2bf(kv.z), f2bf(kv.w) };
      *reinterpret_cast<vbf4*>(&Ks[row][c4]) = kp;
      vf4 vv = *reinterpret_cast<const vf4*>(v + g);
      Vt[c4 + 0][row] = f2bf(vv.x);
      Vt[c4 + 1][row] = f2bf(vv.y);
      Vt[c4 + 2][row] = f2bf(vv.z);
      Vt[c4 + 3][row] = f2bf(vv.w);
    }
    __syncthreads();

    // ---- S = Q @ K^T  (two 16x16 tiles over 32 keys)
    v8f sacc[2];
    sacc[0] = vzero; sacc[1] = vzero;
    #pragma unroll
    for (int nt = 0; nt < 2; ++nt) {
      int col = nt * 16 + lo;                // key column
      #pragma unroll
      for (int c = 0; c < 4; ++c) {
        Frag bfK;
        int db = c * 32 + hi * 8;
        bfK.f4[0] = *reinterpret_cast<const vf4*>(&Ks[col][db]);
        bfK.f4[1] = *reinterpret_cast<const vf4*>(&Ks[col][db + 16]);
        sacc[nt] = __builtin_amdgcn_wmma_f32_16x16x32_bf16(
            false, qf[c].v, false, bfK.v, (short)0, sacc[nt], false, false);
      }
    }

    // ---- online softmax (rows live per-VGPR-index across 16-lane halves)
    float alpha[8];
    #pragma unroll
    for (int r = 0; r < 8; ++r) {
      float s0 = sacc[0][r] * scale2;
      float s1 = sacc[1][r] * scale2;
      float mx = fmaxf(s0, s1);
      #pragma unroll
      for (int m = 1; m < 16; m <<= 1) mx = fmaxf(mx, __shfl_xor(mx, m, 32));
      float mnew = fmaxf(mrow[r], mx);
      alpha[r] = fast_exp2(mrow[r] - mnew);
      float p0 = fast_exp2(s0 - mnew);
      float p1 = fast_exp2(s1 - mnew);
      float ps = p0 + p1;
      #pragma unroll
      for (int m = 1; m < 16; m <<= 1) ps += __shfl_xor(ps, m, 32);
      lrow[r] = lrow[r] * alpha[r] + ps;
      mrow[r] = mnew;
      Ps[wave][hi * 8 + r][lo]      = f2bf(p0);
      Ps[wave][hi * 8 + r][16 + lo] = f2bf(p1);
    }

    // ---- rescale O, then O += P @ V
    #pragma unroll
    for (int t = 0; t < 8; ++t)
      #pragma unroll
      for (int r = 0; r < 8; ++r) oacc[t][r] *= alpha[r];

    Frag pf;
    pf.f4[0] = *reinterpret_cast<const vf4*>(&Ps[wave][lo][hi * 8]);
    pf.f4[1] = *reinterpret_cast<const vf4*>(&Ps[wave][lo][hi * 8 + 16]);
    #pragma unroll
    for (int t = 0; t < 8; ++t) {
      Frag vfB;
      int col = t * 16 + lo;                 // output d column
      int kk  = hi * 8;
      vfB.f4[0] = *reinterpret_cast<const vf4*>(&Vt[col][kk]);
      vfB.f4[1] = *reinterpret_cast<const vf4*>(&Vt[col][kk + 16]);
      oacc[t] = __builtin_amdgcn_wmma_f32_16x16x32_bf16(
          false, pf.v, false, vfB.v, (short)0, oacc[t], false, false);
    }
    __syncthreads();
  }

  // ---- finalize: O /= l, write [B,S,H,D]
  #pragma unroll
  for (int r = 0; r < 8; ++r) {
    float inv = 1.0f / lrow[r];
    int row = q0 + wave * 16 + hi * 8 + r;
    float* op = o + bh_base + (size_t)row * srow;
    #pragma unroll
    for (int t = 0; t < 8; ++t)
      op[t * 16 + lo] = oacc[t][r] * inv;
  }
}

// ---------------------------------------------------------------------------
// Launch
// ---------------------------------------------------------------------------
extern "C" void kernel_launch(void* const* d_in, const int* in_sizes, int n_in,
                              void* d_out, int out_size, void* d_ws, size_t ws_size,
                              hipStream_t stream) {
  const float* x   = (const float*)d_in[0];
  const int*   pid = (const int*)  d_in[1];
  const float* Wq  = (const float*)d_in[2];
  const float* bq  = (const float*)d_in[3];
  const float* Wk  = (const float*)d_in[4];
  const float* bk  = (const float*)d_in[5];
  const float* Wv  = (const float*)d_in[6];
  const float* bv  = (const float*)d_in[7];
  const float* Wo  = (const float*)d_in[8];
  const float* bo  = (const float*)d_in[9];
  float* out = (float*)d_out;

  const int M = BATCH * SEQ;     // 4096
  const int N = EMBED;           // 2048
  const int K = EMBED;           // 2048
  size_t sz = (size_t)M * N;     // elements per buffer

  float* qb = (float*)d_ws;
  float* kb = qb + sz;
  float* vb = kb + sz;
  float* ab = vb + sz;

  dim3 gg(N / 256, M / 128);     // (8, 32)
  gemm_bias_kernel<<<gg, 256, 0, stream>>>(x, Wq, bq, qb, M, N, K);
  gemm_bias_kernel<<<gg, 256, 0, stream>>>(x, Wk, bk, kb, M, N, K);
  gemm_bias_kernel<<<gg, 256, 0, stream>>>(x, Wv, bv, vb, M, N, K);

  int rope_total = BATCH * SEQ * NHEAD * HALF;   // 4,194,304
  rope_kernel<<<rope_total / 256, 256, 0, stream>>>(qb, kb, pid);

  dim3 ga(SEQ / 64, BATCH * NHEAD);              // (32, 32)
  flash_attn_kernel<<<ga, 128, 0, stream>>>(qb, kb, vb, ab);

  gemm_bias_kernel<<<gg, 256, 0, stream>>>(ab, Wo, bo, out, M, N, K);
}
